// LittleBitLinear_81982335746212
// MI455X (gfx1250) — matrix operand order repaired and checked by reading, not compiled, not run on hardware
//
#include <hip/hip_runtime.h>
#include <hip/hip_bf16.h>

typedef __attribute__((ext_vector_type(16))) __bf16 v16bf;
typedef __attribute__((ext_vector_type(8)))  __bf16 v8bf;
typedef __attribute__((ext_vector_type(4)))  __bf16 v4bf;
typedef __attribute__((ext_vector_type(8)))  float  v8f;
typedef int v4i __attribute__((vector_size(16)));   // matches async-to-LDS builtin param

#define TOKENS 8192
#define D_IN   4096
#define D_OUT  4096
#define RANK   1024
#define RANK2  2048   // both pathways concatenated

#define GLOBAL_AS __attribute__((address_space(1)))
#define LDS_AS    __attribute__((address_space(3)))

#if __has_builtin(__builtin_amdgcn_global_load_async_to_lds_b128) && \
    __has_builtin(__builtin_amdgcn_s_wait_asynccnt)
#define USE_ASYNC_LDS 1
#else
#define USE_ASYNC_LDS 0
#endif

__device__ __forceinline__ float signf(float v) {
    return (v > 0.f) ? 1.f : ((v < 0.f) ? -1.f : 0.f);
}

// ---------------- prep kernels ----------------

__global__ void k_cvt_x(const float* __restrict__ x, __bf16* __restrict__ xb) {
    int i = (blockIdx.x * blockDim.x + threadIdx.x) * 4;
    float4 v = *(const float4*)(x + i);
    v4bf o;
    o[0] = (__bf16)v.x; o[1] = (__bf16)v.y; o[2] = (__bf16)v.z; o[3] = (__bf16)v.w;
    *(v4bf*)(xb + i) = o;
}

// W1t[n][k] = g[k] * sign(V[k][n%1024]), n-major, K contiguous
__global__ void k_build_w1t(const float* __restrict__ Vp, const float* __restrict__ Vr,
                            const float* __restrict__ gp, const float* __restrict__ gr,
                            __bf16* __restrict__ w1t) {
    int idx = (blockIdx.x * blockDim.x + threadIdx.x) * 4;   // over 2048*4096
    int n = idx >> 12;
    int k = idx & (D_IN - 1);
    const float* V = (n < RANK) ? Vp : Vr;
    const float* g = (n < RANK) ? gp : gr;
    int nn = n & (RANK - 1);
    v4bf o;
#pragma unroll
    for (int j = 0; j < 4; ++j) {
        float v = V[(size_t)(k + j) * RANK + nn];
        o[j] = (__bf16)(g[k + j] * signf(v));
    }
    *(v4bf*)(w1t + idx) = o;
}

// W2t[d][r] = l[r%1024] * h[d] * sign(U[d][r%1024]), d-major, R contiguous
__global__ void k_build_w2t(const float* __restrict__ Up, const float* __restrict__ Ur,
                            const float* __restrict__ lp, const float* __restrict__ lr,
                            const float* __restrict__ hp, const float* __restrict__ hr,
                            __bf16* __restrict__ w2t) {
    int idx = (blockIdx.x * blockDim.x + threadIdx.x) * 4;   // over 4096*2048
    int d = idx >> 11;
    int r = idx & (RANK2 - 1);
    const float* U = (r < RANK) ? Up : Ur;
    const float* l = (r < RANK) ? lp : lr;
    const float* h = (r < RANK) ? hp : hr;
    int rr = r & (RANK - 1);
    float hd = h[d];
    v4bf o;
#pragma unroll
    for (int j = 0; j < 4; ++j) {
        float u = U[(size_t)d * RANK + rr + j];
        o[j] = (__bf16)(l[rr + j] * hd * signf(u));
    }
    *(v4bf*)(w2t + idx) = o;
}

// ---------------- WMMA GEMM ----------------
// C[M,N] = A[M,K] @ Bt[N,K]^T  (bf16 operands K-major, fp32 accumulate)
// 256 threads = 8 waves; block tile 128x128; K-step 64, double-buffered LDS
// staged via GLOBAL_LOAD_ASYNC_TO_LDS_B128 (ASYNCcnt). Waves 4(M) x 2(N),
// each wave owns 32x64 = 2x4 WMMA accumulators. Last K-iteration peeled so
// the steady-state loop is branch-free: issue-next / s_wait_asynccnt 8 /
// barrier / 16 WMMAs / barrier.
union Frag { v16bf v; v8bf h[2]; };

template<bool FINAL>
__global__ __launch_bounds__(256)
void k_gemm(const __bf16* __restrict__ A, const __bf16* __restrict__ Bt,
            float* __restrict__ Cf, __bf16* __restrict__ Cb,
            const float* __restrict__ bias, int M, int N, int K) {
    constexpr int LDT  = 72;                      // 64 + 8 pad: stride 144B = 36 banks
    constexpr int TILE = 128 * LDT;               // one buffer, bf16 elems
    __shared__ __attribute__((aligned(16))) __bf16 As[2 * TILE];
    __shared__ __attribute__((aligned(16))) __bf16 Bs[2 * TILE];

    const int tid   = threadIdx.x;
    const int lane  = tid & 31;
    const int wave  = tid >> 5;
    const int waveM = wave & 3;                   // 4 waves * 32 rows = 128
    const int waveN = wave >> 2;                  // 2 waves * 64 cols = 128
    const int half  = lane >> 4;
    const int lmod  = lane & 15;

    // staging map: 1024 chunks of 16B per 128x64 tile; thread t owns chunks t+j*256
    size_t gA[4], gB[4];
    int    lofs[4];
#pragma unroll
    for (int j = 0; j < 4; ++j) {
        int c   = tid + j * 256;
        int row = c >> 2;                          // 0..255
        int q   = c & 3;                           // 16B chunk within 64-bf16 row
        gA[j]   = (size_t)(blockIdx.y * 128 + row) * K + q * 8;
        gB[j]   = (size_t)(blockIdx.x * 128 + row) * K + q * 8;
        lofs[j] = row * LDT + q * 8;
    }

    v8f acc[2][4] = {};
    const int nK = K >> 6;                        // K-steps of 64

    // issue one 128x64 A-tile + 128x64 B-tile (8 async b128 per wave)
    auto issue = [&](int k0, int bufb) {
#if USE_ASYNC_LDS
#pragma unroll
        for (int j = 0; j < 4; ++j)
            __builtin_amdgcn_global_load_async_to_lds_b128(
                (GLOBAL_AS v4i*)(A + gA[j] + k0),
                (LDS_AS v4i*)(&As[bufb + lofs[j]]), 0, 0);
#pragma unroll
        for (int j = 0; j < 4; ++j)
            __builtin_amdgcn_global_load_async_to_lds_b128(
                (GLOBAL_AS v4i*)(Bt + gB[j] + k0),
                (LDS_AS v4i*)(&Bs[bufb + lofs[j]]), 0, 0);
#endif
    };

    // two 32-deep WMMA slices from buffer at bf16 offset curb
    auto compute = [&](int curb) {
#pragma unroll
        for (int ks = 0; ks < 2; ++ks) {
            Frag fa[2];
#pragma unroll
            for (int tm = 0; tm < 2; ++tm) {
                int m = waveM * 32 + tm * 16 + lmod;
                fa[tm].h[0] = *(const v8bf*)(&As[curb + m * LDT + ks * 32 + half * 8]);
                fa[tm].h[1] = *(const v8bf*)(&As[curb + m * LDT + ks * 32 + 16 + half * 8]);
            }
            Frag fb[4];
#pragma unroll
            for (int tn = 0; tn < 4; ++tn) {
                int n = waveN * 64 + tn * 16 + lmod;
                fb[tn].h[0] = *(const v8bf*)(&Bs[curb + n * LDT + ks * 32 + half * 8]);
                fb[tn].h[1] = *(const v8bf*)(&Bs[curb + n * LDT + ks * 32 + 16 + half * 8]);
            }
#pragma unroll
            for (int tm = 0; tm < 2; ++tm)
#pragma unroll
                for (int tn = 0; tn < 4; ++tn)
                    acc[tm][tn] = __builtin_amdgcn_wmma_f32_16x16x32_bf16(
                        false, fa[tm].v, false, fb[tn].v,
                        (short)0, acc[tm][tn], false, false);
        }
    };

#if USE_ASYNC_LDS
    issue(0, 0);                                   // prologue: stage 0 -> buffer 0
    for (int kt = 0; kt < nK - 1; ++kt) {          // steady state, branch-free body
        const int cur = kt & 1;
        issue((kt + 1) << 6, (cur ^ 1) * TILE);    // prefetch stage kt+1
        __builtin_amdgcn_s_wait_asynccnt(8);       // stage kt landed (in-order)
        __syncthreads();                           // visible to all waves
        compute(cur * TILE);
        __syncthreads();                           // WAR: buffer reuse guard
    }
    {                                              // peeled last iteration
        const int cur = (nK - 1) & 1;
        __builtin_amdgcn_s_wait_asynccnt(0);
        __syncthreads();
        compute(cur * TILE);
    }
#else
    // fallback: synchronous VGPR staging
    for (int kt = 0; kt < nK; ++kt) {
        const int k0   = kt << 6;
        const int curb = (kt & 1) * TILE;
        float4 ra[4], rb[4];
#pragma unroll
        for (int j = 0; j < 4; ++j) {
            ra[j] = *(const float4*)(A + gA[j] + k0);
            rb[j] = *(const float4*)(Bt + gB[j] + k0);
        }
        __syncthreads();
#pragma unroll
        for (int j = 0; j < 4; ++j) {
            *(float4*)(&As[curb + lofs[j]]) = ra[j];
            *(float4*)(&Bs[curb + lofs[j]]) = rb[j];
        }
        __syncthreads();
        compute(curb);
    }
#endif

    // ---- epilogue ----  C layout: elem i -> row = i + 8*half, col = lmod
#pragma unroll
    for (int tm = 0; tm < 2; ++tm) {
#pragma unroll
        for (int tn = 0; tn < 4; ++tn) {
            int r0 = blockIdx.y * 128 + waveM * 32 + tm * 16 + half * 8;
            int c  = blockIdx.x * 128 + waveN * 64 + tn * 16 + lmod;
            if (FINAL) {
                float bv = bias[c];
#pragma unroll
                for (int i = 0; i < 8; ++i)
                    Cf[(size_t)(r0 + i) * N + c] = acc[tm][tn][i] + bv;
            } else {
#pragma unroll
                for (int i = 0; i < 8; ++i)
                    Cb[(size_t)(r0 + i) * N + c] = (__bf16)acc[tm][tn][i];
            }
        }
    }
}

// ---------------- launch ----------------
extern "C" void kernel_launch(void* const* d_in, const int* in_sizes, int n_in,
                              void* d_out, int out_size, void* d_ws, size_t ws_size,
                              hipStream_t stream) {
    const float* x    = (const float*)d_in[0];
    const float* Up   = (const float*)d_in[1];
    const float* Vp   = (const float*)d_in[2];
    const float* hp   = (const float*)d_in[3];
    const float* lp   = (const float*)d_in[4];
    const float* gp   = (const float*)d_in[5];
    const float* Ur   = (const float*)d_in[6];
    const float* Vr   = (const float*)d_in[7];
    const float* hr   = (const float*)d_in[8];
    const float* lr   = (const float*)d_in[9];
    const float* gr   = (const float*)d_in[10];
    const float* bias = (const float*)d_in[11];

    char* ws = (char*)d_ws;
    __bf16* xb  = (__bf16*)(ws);                          // 64 MB: [8192,4096]
    __bf16* w1t = (__bf16*)(ws + (size_t)(64u  << 20));   // 16 MB: [2048,4096] K-major
    __bf16* w2t = (__bf16*)(ws + (size_t)(80u  << 20));   // 16 MB: [4096,2048] K-major
    __bf16* mid = (__bf16*)(ws + (size_t)(96u  << 20));   // 32 MB: [8192,2048]

    k_cvt_x    <<<(TOKENS * D_IN) / (256 * 4), 256, 0, stream>>>(x, xb);
    k_build_w1t<<<(RANK2 * D_IN) / (256 * 4), 256, 0, stream>>>(Vp, Vr, gp, gr, w1t);
    k_build_w2t<<<(D_OUT * RANK2) / (256 * 4), 256, 0, stream>>>(Up, Ur, lp, lr, hp, hr, w2t);

    // mid[8192,2048] = xb @ W1cat   (K = 4096)
    k_gemm<false><<<dim3(RANK2 / 128, TOKENS / 128), 256, 0, stream>>>(
        xb, w1t, nullptr, mid, nullptr, TOKENS, RANK2, D_IN);

    // out[8192,4096] = mid @ W2cat + bias   (K = 2048; l,h folded into W2cat)
    k_gemm<true><<<dim3(D_OUT / 128, TOKENS / 128), 256, 0, stream>>>(
        mid, w2t, (float*)d_out, nullptr, bias, TOKENS, D_OUT, RANK2);
}
